// PaliGemmaWithExpertAndGripperTactileModel_15040975471223
// MI455X (gfx1250) — matrix operand order repaired and checked by reading, not compiled, not run on hardware
//
#include <hip/hip_runtime.h>
#include <hip/hip_bf16.h>
#include <stdint.h>

// ---------------- model constants ----------------
#define LNUM   4
#define BB     2
#define SEQ    882
#define SEQP   896          // SEQ padded to a multiple of 32 (and of 64/128 tiles)
#define NH     8
#define HDIM   256
#define DMODEL 2048

typedef __attribute__((ext_vector_type(16))) __bf16    v16bf;
typedef __attribute__((ext_vector_type(2)))  __bf16    v2bf;
typedef __attribute__((ext_vector_type(8)))  float     v8f;
typedef __attribute__((ext_vector_type(4)))  unsigned  u32x4;

struct U8 { u32x4 a, b; };   // 32B, bit-castable to v16bf

// native f32 -> bf16 conversion (hardware cvt, RTNE)
__device__ __forceinline__ unsigned short f2bf_u16(float f) {
  return __builtin_bit_cast(unsigned short, (__bf16)f);
}

__device__ __forceinline__ unsigned pack2bf(float a, float b) {
  v2bf v = {(__bf16)a, (__bf16)b};
  return __builtin_bit_cast(unsigned, v);
}

__device__ __forceinline__ v16bf make_frag(const unsigned short* p0,
                                           const unsigned short* p1) {
  U8 t;
  t.a = *(const u32x4*)p0;
  t.b = *(const u32x4*)p1;
  return __builtin_bit_cast(v16bf, t);
}

// row remap: logical GEMM row r -> global row (r/len)*stride + off + r%len
struct Map { int len; long long stride; long long off; };

struct GemmP {
  const float* A;  long long Az1, Az2; int lda; Map am;
  const float* Bm; long long Bz1, Bz2; int ldb;
  float*       C;  long long Cz1, Cz2; int ldc; Map cm;
  const float* R;  int ldr;            // optional residual (uses cm)
  int M, N, K, zdiv;
};

// ---------------------------------------------------------------
// bf16 WMMA GEMM: C = A * B (+R).  Block tile 64x128, 8 waves of
// 32x32, K-step 32.  fp32 tiles converted to bf16 while staged into
// LDS in exactly the VGPR fragment layout of v_wmma 16x16x32.
// PRECONDITIONS (guaranteed by the launcher):  K % 32 == 0 and
// N % 128 == 0, so only an M-row guard remains (act expert, M=100).
// BT=true means B is given as [N,K] row-major (NT GEMM, q*k^T).
// ---------------------------------------------------------------
template <bool BT>
__global__ __launch_bounds__(256) void gemm_bf16_wmma(GemmP p) {
  __shared__ alignas(16) unsigned short As[64][40];    // [m][k], pad 40
  __shared__ alignas(16) unsigned short Bs[128][40];   // [n][k] (transposed)

  const int tid  = threadIdx.x;
  const int z    = blockIdx.z;
  const long long aoff = (long long)(z / p.zdiv) * p.Az1 + (long long)(z % p.zdiv) * p.Az2;
  const long long boff = (long long)(z / p.zdiv) * p.Bz1 + (long long)(z % p.zdiv) * p.Bz2;
  const long long coff = (long long)(z / p.zdiv) * p.Cz1 + (long long)(z % p.zdiv) * p.Cz2;
  const int m0 = blockIdx.y * 64;
  const int n0 = blockIdx.x * 128;

  const int w    = tid >> 5, lane = tid & 31;
  const int wm   = w & 1,    wn   = w >> 1;       // 2 x 4 wave grid
  const int half = lane >> 4, l15 = lane & 15;

  const bool mFull = (m0 + 64 <= p.M);

  v8f acc[2][2] = {};

  // A staging geometry: thread -> (row, 8 consecutive k)
  const int arow = tid >> 2;            // 0..63
  const int akb  = (tid & 3) * 8;       // 0,8,16,24
  const int ar   = m0 + arow;
  const int arcl = ar < p.M ? ar : (p.M - 1);
  const long long agr = (long long)(arcl / p.am.len) * p.am.stride + p.am.off + (arcl % p.am.len);
  const float* asrc = p.A + aoff + agr * p.lda;
  const bool arOK = (ar < p.M);

  // B staging geometry (no guards needed)
  const int bk  = tid >> 3;             // !BT: 0..31  (row k)
  const int bnb = (tid & 7) * 16;       // !BT: 16 n per thread
  const int btn = tid >> 1;             // BT: 0..127  (row n)
  const int btk = (tid & 1) * 16;       // BT: 16 consecutive k

  for (int kt = 0; kt < p.K; kt += 32) {
    // ---------------- stage A ----------------
    if (mFull) {
      const float4* s4 = (const float4*)(asrc + kt + akb);
      float4 f0 = s4[0], f1 = s4[1];
      u32x4 pk;
      pk.x = pack2bf(f0.x, f0.y);
      pk.y = pack2bf(f0.z, f0.w);
      pk.z = pack2bf(f1.x, f1.y);
      pk.w = pack2bf(f1.z, f1.w);
      *(u32x4*)&As[arow][akb] = pk;
      if (kt + 32 < p.K) __builtin_prefetch(asrc + kt + 32 + akb, 0, 1);
    } else {
      u32x4 pk = {0u, 0u, 0u, 0u};
      if (arOK) {
        const float4* s4 = (const float4*)(asrc + kt + akb);
        float4 f0 = s4[0], f1 = s4[1];
        pk.x = pack2bf(f0.x, f0.y);
        pk.y = pack2bf(f0.z, f0.w);
        pk.z = pack2bf(f1.x, f1.y);
        pk.w = pack2bf(f1.z, f1.w);
      }
      *(u32x4*)&As[arow][akb] = pk;
    }

    // ---------------- stage B (transposed in LDS: Bs[n][k]) ----------------
    if (!BT) {
      const float* bsrc = p.Bm + boff + (long long)(kt + bk) * p.ldb + n0 + bnb;
      #pragma unroll
      for (int j = 0; j < 4; ++j) {
        float4 f = ((const float4*)bsrc)[j];
        Bs[bnb + 4 * j + 0][bk] = f2bf_u16(f.x);
        Bs[bnb + 4 * j + 1][bk] = f2bf_u16(f.y);
        Bs[bnb + 4 * j + 2][bk] = f2bf_u16(f.z);
        Bs[bnb + 4 * j + 3][bk] = f2bf_u16(f.w);
      }
      if (kt + 32 < p.K) __builtin_prefetch(bsrc + 32LL * p.ldb, 0, 1);
    } else {
      const float* bsrc = p.Bm + boff + (long long)(n0 + btn) * p.ldb + kt + btk;
      float4 f0 = ((const float4*)bsrc)[0];
      float4 f1 = ((const float4*)bsrc)[1];
      float4 f2 = ((const float4*)bsrc)[2];
      float4 f3 = ((const float4*)bsrc)[3];
      u32x4 lo, hi;
      lo.x = pack2bf(f0.x, f0.y); lo.y = pack2bf(f0.z, f0.w);
      lo.z = pack2bf(f1.x, f1.y); lo.w = pack2bf(f1.z, f1.w);
      hi.x = pack2bf(f2.x, f2.y); hi.y = pack2bf(f2.z, f2.w);
      hi.z = pack2bf(f3.x, f3.y); hi.w = pack2bf(f3.z, f3.w);
      *(u32x4*)&Bs[btn][btk]     = lo;
      *(u32x4*)&Bs[btn][btk + 8] = hi;
      if (kt + 32 < p.K) __builtin_prefetch(bsrc + 32, 0, 1);
    }
    __syncthreads();

    // ---- fragments per ISA 7.12.2 ----
    v16bf afr[2], bfr[2];
    #pragma unroll
    for (int mi = 0; mi < 2; ++mi) {
      const unsigned short* ap = &As[wm * 32 + mi * 16 + l15][half * 8];
      afr[mi] = make_frag(ap, ap + 16);   // K k0..k0+7 then k0+16..k0+23
    }
    #pragma unroll
    for (int ni = 0; ni < 2; ++ni) {
      const unsigned short* bp = &Bs[wn * 32 + ni * 16 + l15][half * 8];
      bfr[ni] = make_frag(bp, bp + 16);   // K k0..k0+7 then k0+16..k0+23
    }
    #pragma unroll
    for (int mi = 0; mi < 2; ++mi)
      #pragma unroll
      for (int ni = 0; ni < 2; ++ni)
        acc[mi][ni] = __builtin_amdgcn_wmma_f32_16x16x32_bf16(
            false, afr[mi], false, bfr[ni], (short)0, acc[mi][ni], false, false);
    __syncthreads();
  }

  // ---- epilogue: C VGPR i -> row i (+8 for hi half-wave), col = lane&15 ----
  #pragma unroll
  for (int mi = 0; mi < 2; ++mi)
    #pragma unroll
    for (int i = 0; i < 8; ++i) {
      int r = m0 + wm * 32 + mi * 16 + i + (half ? 8 : 0);
      if (!mFull && r >= p.M) continue;
      long long gr = (long long)(r / p.cm.len) * p.cm.stride + p.cm.off + (r % p.cm.len);
      float*       crow = p.C + coff + gr * p.ldc;
      const float* rrow = p.R ? (p.R + gr * p.ldr) : nullptr;
      #pragma unroll
      for (int ni = 0; ni < 2; ++ni) {
        int c = n0 + wn * 32 + ni * 16 + l15;
        float v = acc[mi][ni][i];
        if (rrow) v += rrow[c];
        crow[c] = v;
      }
    }
}

// ---------------- rmsnorm: one block per row ----------------
__global__ __launch_bounds__(256) void rmsnorm_k(const float* __restrict__ x,
                                                 const float* __restrict__ w,
                                                 float* __restrict__ out, int W) {
  long long row = blockIdx.x;
  const float* xr = x + row * W;
  float ss = 0.f;
  for (int c = threadIdx.x; c < W; c += 256) { float v = xr[c]; ss += v * v; }
  for (int m = 16; m; m >>= 1) ss += __shfl_xor(ss, m, 32);
  __shared__ float red[8];
  if ((threadIdx.x & 31) == 0) red[threadIdx.x >> 5] = ss;
  __syncthreads();
  float tot = 0.f;
  #pragma unroll
  for (int i = 0; i < 8; ++i) tot += red[i];
  float sc = rsqrtf(tot / (float)W + 1e-6f);
  float* orow = out + row * W;
  for (int c = threadIdx.x; c < W; c += 256) orow[c] = xr[c] * sc * (1.f + w[c]);
}

// ---------------- rope: block per (s,b), 128 threads = dim pairs ----------------
__global__ void rope_k(float* __restrict__ q, float* __restrict__ k,
                       const int* __restrict__ pos, int S) {
  int s = blockIdx.x, b = blockIdx.y, t = threadIdx.x;   // t in [0,128)
  float p   = (float)pos[b * S + s];
  float inv = __expf(-(float)(2 * t) / (float)HDIM * 9.210340371976184f); // ln(1e4)
  float fr  = p * inv;
  float cs = __cosf(fr), sn = __sinf(fr);
  long long kb = ((long long)b * SEQP + s) * HDIM;
  float a0 = k[kb + t], a1 = k[kb + t + 128];
  k[kb + t]       = a0 * cs - a1 * sn;
  k[kb + t + 128] = a1 * cs + a0 * sn;
  long long qb = ((long long)b * SEQP + s) * DMODEL;
  #pragma unroll
  for (int h = 0; h < NH; ++h) {
    long long o = qb + (long long)h * HDIM;
    float x0 = q[o + t], x1 = q[o + t + 128];
    q[o + t]       = x0 * cs - x1 * sn;
    q[o + t + 128] = x1 * cs + x0 * sn;
  }
}

// ------- softmax over a row; row pulled into LDS with CDNA5 async copy -------
// Rows live in a (B, H, SEQP, SEQP) buffer; only the first SEQ columns are
// real.  Pad columns are written as exact zeros so the padded-K probs@v GEMM
// is unaffected by the padding.
__global__ __launch_bounds__(256) void softmax_k(float* __restrict__ scores,
                                                 const float* __restrict__ mask,
                                                 float scale) {
  long long rid = blockIdx.x;                      // over B*H*SEQ (real rows)
  int qpos = (int)(rid % SEQ);
  int bh   = (int)(rid / SEQ);                     // b*H + h
  int b    = bh / NH;
  float* sr = scores + ((long long)bh * SEQP + qpos) * SEQP;
  const float* mr = mask + ((long long)b * SEQ + qpos) * SEQ;

  __shared__ alignas(16) float buf[896];
  __shared__ float red[8];
  const int tid = threadIdx.x;

  // async global -> LDS copy of the row (ASYNCcnt path)
  for (int i = tid; i < SEQ; i += 256) {
    unsigned lo = (unsigned)(size_t)&buf[i];
    asm volatile("global_load_async_to_lds_b32 %0, %1, %2"
                 :: "v"(lo), "v"(i * 4), "s"(sr) : "memory");
  }
  asm volatile("s_wait_asynccnt 0" ::: "memory");
  __syncthreads();

  float mx = -3.4e38f;
  for (int i = tid; i < SEQ; i += 256) {
    float v = buf[i] * scale + mr[i];
    buf[i] = v;
    mx = fmaxf(mx, v);
  }
  for (int m = 16; m; m >>= 1) mx = fmaxf(mx, __shfl_xor(mx, m, 32));
  __syncthreads();
  if ((tid & 31) == 0) red[tid >> 5] = mx;
  __syncthreads();
  #pragma unroll
  for (int i = 0; i < 8; ++i) mx = fmaxf(mx, red[i]);

  float sum = 0.f;
  for (int i = tid; i < SEQ; i += 256) { float e = __expf(buf[i] - mx); buf[i] = e; sum += e; }
  for (int m = 16; m; m >>= 1) sum += __shfl_xor(sum, m, 32);
  __syncthreads();
  if ((tid & 31) == 0) red[tid >> 5] = sum;
  __syncthreads();
  float tot = 0.f;
  #pragma unroll
  for (int i = 0; i < 8; ++i) tot += red[i];
  float inv = 1.f / tot;
  for (int i = tid; i < SEQ; i += 256) sr[i] = buf[i] * inv;
  for (int i = SEQ + tid; i < SEQP; i += 256) sr[i] = 0.f;   // K-padding
}

// ---------------- g = gelu_tanh(g) * u ----------------
__global__ void gelumul_k(float* __restrict__ g, const float* __restrict__ u,
                          long long n) {
  long long i = (long long)blockIdx.x * 256 + threadIdx.x;
  if (i < n) {
    float x = g[i];
    float t = tanhf(0.7978845608028654f * (x + 0.044715f * x * x * x));
    g[i] = 0.5f * x * (1.f + t) * u[i];
  }
}

__global__ void copy_k(const float* __restrict__ src, float* __restrict__ dst,
                       long long n) {
  long long i = (long long)blockIdx.x * 256 + threadIdx.x;
  if (i < n) dst[i] = src[i];
}

__global__ void zero_k(float* __restrict__ dst, long long n) {
  long long i = (long long)blockIdx.x * 256 + threadIdx.x;
  if (i < n) dst[i] = 0.f;
}

// ---------------- host-side orchestration ----------------
static GemmP mk(const float* A, int lda, Map am, long long Az1, long long Az2,
                const float* B, int ldb, long long Bz1, long long Bz2,
                float* C, int ldc, Map cm, long long Cz1, long long Cz2,
                const float* R, int ldr, int M, int N, int K, int zdiv) {
  GemmP p;
  p.A = A; p.Az1 = Az1; p.Az2 = Az2; p.lda = lda; p.am = am;
  p.Bm = B; p.Bz1 = Bz1; p.Bz2 = Bz2; p.ldb = ldb;
  p.C = C; p.Cz1 = Cz1; p.Cz2 = Cz2; p.ldc = ldc; p.cm = cm;
  p.R = R; p.ldr = ldr; p.M = M; p.N = N; p.K = K; p.zdiv = zdiv;
  return p;
}

static void run_gemm(hipStream_t st, bool bt, const GemmP& p, int Z) {
  dim3 grid((unsigned)((p.N + 127) / 128), (unsigned)((p.M + 63) / 64), (unsigned)Z);
  if (bt) gemm_bf16_wmma<true><<<grid, 256, 0, st>>>(p);
  else    gemm_bf16_wmma<false><<<grid, 256, 0, st>>>(p);
}

extern "C" void kernel_launch(void* const* d_in, const int* in_sizes, int n_in,
                              void* d_out, int out_size, void* d_ws, size_t ws_size,
                              hipStream_t stream) {
  (void)in_sizes; (void)n_in; (void)out_size; (void)ws_size;
  const int Wd[3] = {2048, 1024, 1024};
  const int Md[3] = {8192, 4096, 4096};
  const int Sl[3] = {768, 64, 50};
  const int So[3] = {0, 768, 832};

  const float* xin[3];
  const float* wt[3][10];   // ln1,q,k,v,o,ln2,gate,up,down,norm
  for (int e = 0; e < 3; ++e) {
    xin[e] = (const float*)d_in[e];
    for (int j = 0; j < 10; ++j) wt[e][j] = (const float*)d_in[5 + e * 10 + j];
  }
  const int*   pos  = (const int*)d_in[3];
  const float* mask = (const float*)d_in[4];
  float*       out  = (float*)d_out;

  // bump allocator in workspace
  uintptr_t wp = (uintptr_t)d_ws;
  auto alloc = [&](size_t elems) -> float* {
    float* r = (float*)wp;
    wp += ((elems * sizeof(float)) + 255) & ~(size_t)255;
    return r;
  };
  float *xA[3], *xB[3], *hb[3], *gb[3], *ub[3];
  for (int e = 0; e < 3; ++e) {
    size_t rw = (size_t)BB * Sl[e] * Wd[e];
    xA[e] = alloc(rw); xB[e] = alloc(rw); hb[e] = alloc(rw);
    size_t rm = (size_t)BB * Sl[e] * Md[e];
    gb[e] = alloc(rm); ub[e] = alloc(rm);
  }
  float* qb  = alloc((size_t)BB * SEQP * DMODEL);
  float* kb  = alloc((size_t)BB * SEQP * HDIM);
  float* vb  = alloc((size_t)BB * SEQP * HDIM);
  float* sc  = alloc((size_t)BB * NH * SEQP * SEQP);
  float* att = alloc((size_t)BB * SEQP * DMODEL);

  const Map id = {1 << 30, 0, 0};
  const float scale = 0.0625f;   // HD^-0.5

  // residual init + zero the sequence-padding rows of q/k/v (done once;
  // the projection GEMMs only ever write the SEQ real rows)
  for (int e = 0; e < 3; ++e) {
    long long n = (long long)BB * Sl[e] * Wd[e];
    copy_k<<<dim3((unsigned)((n + 255) / 256)), 256, 0, stream>>>(xin[e], xA[e], n);
  }
  for (int b = 0; b < BB; ++b) {
    long long padq = (long long)(SEQP - SEQ) * DMODEL;
    long long padk = (long long)(SEQP - SEQ) * HDIM;
    zero_k<<<dim3((unsigned)((padq + 255) / 256)), 256, 0, stream>>>(
        qb + ((long long)b * SEQP + SEQ) * DMODEL, padq);
    zero_k<<<dim3((unsigned)((padk + 255) / 256)), 256, 0, stream>>>(
        kb + ((long long)b * SEQP + SEQ) * HDIM, padk);
    zero_k<<<dim3((unsigned)((padk + 255) / 256)), 256, 0, stream>>>(
        vb + ((long long)b * SEQP + SEQ) * HDIM, padk);
  }

  for (int l = 0; l < LNUM; ++l) {
    // ln1
    for (int e = 0; e < 3; ++e)
      rmsnorm_k<<<BB * Sl[e], 256, 0, stream>>>(
          xA[e], wt[e][0] + (size_t)l * Wd[e], hb[e], Wd[e]);

    // q/k/v projections, packed into unified (B,SEQP,*) buffers
    for (int e = 0; e < 3; ++e) {
      Map cm = {Sl[e], SEQP, So[e]};
      run_gemm(stream, false,
               mk(hb[e], Wd[e], id, 0, 0,
                  wt[e][1] + (size_t)l * Wd[e] * DMODEL, DMODEL, 0, 0,
                  qb, DMODEL, cm, 0, 0, nullptr, 0,
                  BB * Sl[e], DMODEL, Wd[e], 1), 1);
      run_gemm(stream, false,
               mk(hb[e], Wd[e], id, 0, 0,
                  wt[e][2] + (size_t)l * Wd[e] * HDIM, HDIM, 0, 0,
                  kb, HDIM, cm, 0, 0, nullptr, 0,
                  BB * Sl[e], HDIM, Wd[e], 1), 1);
      run_gemm(stream, false,
               mk(hb[e], Wd[e], id, 0, 0,
                  wt[e][3] + (size_t)l * Wd[e] * HDIM, HDIM, 0, 0,
                  vb, HDIM, cm, 0, 0, nullptr, 0,
                  BB * Sl[e], HDIM, Wd[e], 1), 1);
    }

    rope_k<<<dim3(SEQ, BB), 128, 0, stream>>>(qb, kb, pos, SEQ);

    // scores = q * k^T   (z over B*H; M=N=SEQP, all tiles full)
    run_gemm(stream, true,
             mk(qb, DMODEL, id, (long long)SEQP * DMODEL, HDIM,
                kb, HDIM, (long long)SEQP * HDIM, 0,
                sc, SEQP, id, (long long)NH * SEQP * SEQP, (long long)SEQP * SEQP,
                nullptr, 0, SEQP, SEQP, HDIM, NH), BB * NH);

    softmax_k<<<BB * NH * SEQ, 256, 0, stream>>>(sc, mask, scale);

    // att = probs * v  (K=SEQP, pad columns of probs are exact zeros)
    run_gemm(stream, false,
             mk(sc, SEQP, id, (long long)NH * SEQP * SEQP, (long long)SEQP * SEQP,
                vb, HDIM, (long long)SEQP * HDIM, 0,
                att, DMODEL, id, (long long)SEQP * DMODEL, HDIM,
                nullptr, 0, SEQP, HDIM, SEQP, NH), BB * NH);

    // o projection + residual
    for (int e = 0; e < 3; ++e) {
      Map am = {Sl[e], SEQP, So[e]};
      run_gemm(stream, false,
               mk(att, DMODEL, am, 0, 0,
                  wt[e][4] + (size_t)l * DMODEL * Wd[e], Wd[e], 0, 0,
                  xB[e], Wd[e], id, 0, 0, xA[e], Wd[e],
                  BB * Sl[e], Wd[e], DMODEL, 1), 1);
    }

    // ln2 + MLP
    for (int e = 0; e < 3; ++e)
      rmsnorm_k<<<BB * Sl[e], 256, 0, stream>>>(
          xB[e], wt[e][5] + (size_t)l * Wd[e], hb[e], Wd[e]);

    for (int e = 0; e < 3; ++e) {
      run_gemm(stream, false,
               mk(hb[e], Wd[e], id, 0, 0,
                  wt[e][6] + (size_t)l * Wd[e] * Md[e], Md[e], 0, 0,
                  gb[e], Md[e], id, 0, 0, nullptr, 0,
                  BB * Sl[e], Md[e], Wd[e], 1), 1);
      run_gemm(stream, false,
               mk(hb[e], Wd[e], id, 0, 0,
                  wt[e][7] + (size_t)l * Wd[e] * Md[e], Md[e], 0, 0,
                  ub[e], Md[e], id, 0, 0, nullptr, 0,
                  BB * Sl[e], Md[e], Wd[e], 1), 1);
      long long n = (long long)BB * Sl[e] * Md[e];
      gelumul_k<<<dim3((unsigned)((n + 255) / 256)), 256, 0, stream>>>(gb[e], ub[e], n);
      run_gemm(stream, false,
               mk(gb[e], Md[e], id, 0, 0,
                  wt[e][8] + (size_t)l * Md[e] * Wd[e], Wd[e], 0, 0,
                  xA[e], Wd[e], id, 0, 0, xB[e], Wd[e],
                  BB * Sl[e], Wd[e], Md[e], 1), 1);
    }
  }

  // final norms into concatenated output
  long long ooff = 0;
  for (int e = 0; e < 3; ++e) {
    rmsnorm_k<<<BB * Sl[e], 256, 0, stream>>>(xA[e], wt[e][9], out + ooff, Wd[e]);
    ooff += (long long)BB * Sl[e] * Wd[e];
  }
}
